// GCN_19713899888849
// MI455X (gfx1250) — matrix operand order repaired and checked by reading, not compiled, not run on hardware
//
#include <hip/hip_runtime.h>
#include <hip/hip_bf16.h>

typedef __attribute__((ext_vector_type(16))) _Float16 v16h;
typedef __attribute__((ext_vector_type(8)))  float    v8f;

#define NN 50000
#define NE 200000
#define NG 128

// CDNA5 async global->LDS copy path (ASYNCcnt), with sync fallback.
// Builtin param type (from clang diagnostic): v4i in addrspace(1) / LDS ptr.
#if defined(__HIP_DEVICE_COMPILE__) && defined(__gfx1250__) && \
    __has_builtin(__builtin_amdgcn_global_load_async_to_lds_b128) && \
    __has_builtin(__builtin_amdgcn_s_wait_asynccnt)
#define USE_ASYNC_LDS 1
typedef int v4i_vs __attribute__((vector_size(16)));
typedef __attribute__((address_space(1))) v4i_vs as1_v4i;
typedef __attribute__((address_space(3))) v4i_vs as3_v4i;
#define ASYNC_CPY_B128(gsrc, ldst)                                        \
  __builtin_amdgcn_global_load_async_to_lds_b128(                         \
      (as1_v4i*)(void*)(gsrc), (as3_v4i*)(void*)(ldst), 0, 0)
#define ASYNC_WAIT() __builtin_amdgcn_s_wait_asynccnt(0)
#else
#define USE_ASYNC_LDS 0
#define ASYNC_WAIT()
#endif

// ---------------------------------------------------------------------------
// f32 -> f16 convert with zero padding (handles both row and col padding)
// ---------------------------------------------------------------------------
__global__ void k_convert_pad(const float* __restrict__ src, _Float16* __restrict__ dst,
                              int srcR, int srcC, int dstR, int dstC) {
  long long idx = (long long)blockIdx.x * blockDim.x + threadIdx.x;
  long long total = (long long)dstR * dstC;
  if (idx >= total) return;
  int r = (int)(idx / dstC), c = (int)(idx % dstC);
  float v = (r < srcR && c < srcC) ? src[(long long)r * srcC + c] : 0.0f;
  dst[idx] = (_Float16)v;
}

__global__ void k_fill_f32(float* __restrict__ p, float v, long long n) {
  long long idx = (long long)blockIdx.x * blockDim.x + threadIdx.x;
  if (idx < n) p[idx] = v;
}

// ---------------------------------------------------------------------------
// Edge MLP: ew = sigmoid(relu(ea @ W1 + b1) @ W2 + b2), ea: (E,3)
// ---------------------------------------------------------------------------
__global__ void k_edge_mlp(const float* __restrict__ ea, const float* __restrict__ w1,
                           const float* __restrict__ b1, const float* __restrict__ w2,
                           const float* __restrict__ b2, float* __restrict__ ew, int E) {
  int e = blockIdx.x * blockDim.x + threadIdx.x;
  if (e >= E) return;
  float a0 = ea[e * 3 + 0], a1 = ea[e * 3 + 1], a2 = ea[e * 3 + 2];
  float acc = b2[0];
#pragma unroll 8
  for (int j = 0; j < 64; ++j) {
    float t = fmaf(a0, w1[j], fmaf(a1, w1[64 + j], fmaf(a2, w1[128 + j], b1[j])));
    t = fmaxf(t, 0.0f);
    acc = fmaf(t, w2[j], acc);
  }
  ew[e] = 1.0f / (1.0f + __expf(-acc));
}

// ---------------------------------------------------------------------------
// Degree / normalization
// ---------------------------------------------------------------------------
__global__ void k_deg_edges(const int* __restrict__ ei, const float* __restrict__ ew,
                            float* __restrict__ deg, int E) {
  int e = blockIdx.x * blockDim.x + threadIdx.x;
  if (e >= E) return;
  atomicAdd(&deg[ei[E + e]], ew[e]);
}

__global__ void k_rsqrt(const float* __restrict__ deg, float* __restrict__ dis, int n) {
  int i = blockIdx.x * blockDim.x + threadIdx.x;
  if (i < n) dis[i] = rsqrtf(deg[i]);   // deg >= 1 (self-loop) so always > 0
}

// ---------------------------------------------------------------------------
// WMMA GEMM: C[M,N] = op(A[M,K] @ B[K,N] + bias)
// block = 256 threads (8 waves), tile 128x128, wave tile 32x64 (2x4 wmma frags)
// Double-buffered LDS; A tile staged via GLOBAL_LOAD_ASYNC_TO_LDS_B128 when
// available (ASYNCcnt overlap with WMMA), sync loads otherwise.
// ---------------------------------------------------------------------------
template <int OUT_F16, int RELU, int BIAS>
__global__ __launch_bounds__(256) void k_wmma_gemm(
    const _Float16* __restrict__ A, const _Float16* __restrict__ B,
    const float* __restrict__ bias, _Float16* __restrict__ Ch, float* __restrict__ Cf,
    int M, int K, int Nc, int Nstore) {
  __shared__ __attribute__((aligned(16))) _Float16 Als[2][128][32];
  __shared__ __attribute__((aligned(16))) _Float16 Bts[2][128][32];  // [n][k] transposed

  const int tid  = threadIdx.x;
  const int lane = tid & 31;
  const int wave = tid >> 5;
  const int wr = wave >> 1;       // 0..3 : 32-row band inside tile
  const int wc = wave & 1;        // 0..1 : 64-col band inside tile
  const int rowBase = blockIdx.y * 128;
  const int colBase = blockIdx.x * 128;

  v8f acc[2][4] = {};

  // ---- tile staging helpers ----
  auto stageA = [&](int buf, int kb) {
    int r = tid >> 1;
    int cb = (tid & 1) * 16;
    const _Float16* gp = A + (size_t)(rowBase + r) * K + kb + cb;
    if (rowBase + r < M) {
#if USE_ASYNC_LDS
      ASYNC_CPY_B128(gp,     &Als[buf][r][cb]);
      ASYNC_CPY_B128(gp + 8, &Als[buf][r][cb + 8]);
#else
      int4 z0 = ((const int4*)gp)[0];
      int4 z1 = ((const int4*)gp)[1];
      *(int4*)&Als[buf][r][cb] = z0;
      *(int4*)&Als[buf][r][cb + 8] = z1;
#endif
    } else {
      int4 z = {0, 0, 0, 0};
      *(int4*)&Als[buf][r][cb] = z;
      *(int4*)&Als[buf][r][cb + 8] = z;
    }
  };
  auto stageB = [&](int buf, int kb) {
    int kk = tid >> 3;            // 0..31
    int nb = (tid & 7) * 16;      // 0..112
    union { int4 q[2]; _Float16 h[16]; } tb;
    const int4* gp = (const int4*)(B + (size_t)(kb + kk) * Nc + colBase + nb);
    tb.q[0] = gp[0];
    tb.q[1] = gp[1];
#pragma unroll
    for (int j = 0; j < 16; ++j) Bts[buf][nb + j][kk] = tb.h[j];
  };

  stageA(0, 0);
  stageB(0, 0);
  int p = 0;

  for (int kb = 0; kb < K; kb += 32) {
    ASYNC_WAIT();
    __syncthreads();
    if (kb + 32 < K) {            // prefetch next K-step into other buffer
      stageA(p ^ 1, kb + 32);
      stageB(p ^ 1, kb + 32);
    }

    // ---- fragments per documented 16-bit A / B wave32 layouts ----
    const int m  = lane & 15;
    const int kh = lane >> 4;
    union { int4 q[2]; v16h v; } af[2], bf[4];
#pragma unroll
    for (int s = 0; s < 2; ++s) {
      int arow = wr * 32 + s * 16 + m;
      af[s].q[0] = *(const int4*)&Als[p][arow][kh * 8];       // K = kh*8 .. +7
      af[s].q[1] = *(const int4*)&Als[p][arow][16 + kh * 8];  // K = 16+kh*8 .. +7
    }
#pragma unroll
    for (int s = 0; s < 4; ++s) {
      int bcol = wc * 64 + s * 16 + m;                        // n = lane&15
      bf[s].q[0] = *(const int4*)&Bts[p][bcol][kh * 8];       // K = kh*8 .. +7
      bf[s].q[1] = *(const int4*)&Bts[p][bcol][16 + kh * 8];  // K = 16+kh*8 .. +7
    }
#pragma unroll
    for (int i = 0; i < 2; ++i)
#pragma unroll
      for (int j = 0; j < 4; ++j)
        acc[i][j] = __builtin_amdgcn_wmma_f32_16x16x32_f16(
            false, af[i].v, false, bf[j].v, (short)0, acc[i][j], false, false);
    p ^= 1;
  }

  // ---- epilogue: C layout is VGPR r <-> M = r + 8*(lane>=16), N = lane&15 ----
  const int hi = lane >> 4;
  const int n  = lane & 15;
#pragma unroll
  for (int i = 0; i < 2; ++i)
#pragma unroll
    for (int j = 0; j < 4; ++j) {
      int col = colBase + wc * 64 + j * 16 + n;
#pragma unroll
      for (int r = 0; r < 8; ++r) {
        int row = rowBase + wr * 32 + i * 16 + r + 8 * hi;
        if (row < M && col < Nstore) {
          float v = acc[i][j][r];
          if (BIAS) v += bias[col];
          if (RELU) v = fmaxf(v, 0.0f);
          if (OUT_F16) Ch[(size_t)row * Nc + col] = (_Float16)v;
          else         Cf[(size_t)row * Nstore + col] = v;
        }
      }
    }
}

// ---------------------------------------------------------------------------
// GCN scatter stages
// ---------------------------------------------------------------------------
__global__ void k_selfloop(const float* __restrict__ HW, const float* __restrict__ dis,
                           float* __restrict__ Out, int F, long long total) {
  long long idx = (long long)blockIdx.x * blockDim.x + threadIdx.x;
  if (idx >= total) return;
  int row = (int)(idx / F);
  float d = dis[row];
  Out[idx] = d * d * HW[idx];
}

__global__ __launch_bounds__(256) void k_edge_scatter(
    const int* __restrict__ ei, const float* __restrict__ ew, const float* __restrict__ dis,
    const float* __restrict__ HW, float* __restrict__ Out, int E, int F) {
  int e = blockIdx.x;
  __shared__ float sc;
  __shared__ int ss, sd;
  if (threadIdx.x == 0) {
    int s = ei[e], d = ei[E + e];
    ss = s; sd = d;
    sc = dis[s] * ew[e] * dis[d];
  }
  __syncthreads();
  float c = sc;
  const float* hrow = HW + (size_t)ss * F;
  float* orow = Out + (size_t)sd * F;
  for (int f = threadIdx.x; f < F; f += 256)
    atomicAdd(&orow[f], c * hrow[f]);
}

__global__ void k_bias_relu_h(const float* __restrict__ Out, const float* __restrict__ bias,
                              _Float16* __restrict__ H, int F, long long total) {
  long long idx = (long long)blockIdx.x * blockDim.x + threadIdx.x;
  if (idx >= total) return;
  int col = (int)(idx % F);
  H[idx] = (_Float16)fmaxf(Out[idx] + bias[col], 0.0f);
}

// ---------------------------------------------------------------------------
// Pooling
// ---------------------------------------------------------------------------
__global__ void k_pool_add(const _Float16* __restrict__ H, const int* __restrict__ batch,
                           float* __restrict__ sums, int F, long long total) {
  long long idx = (long long)blockIdx.x * blockDim.x + threadIdx.x;
  if (idx >= total) return;
  int row = (int)(idx / F), f = (int)(idx % F);
  atomicAdd(&sums[(size_t)batch[row] * F + f], (float)H[idx]);
}

__global__ void k_count(const int* __restrict__ batch, float* __restrict__ cnt, int n) {
  int i = blockIdx.x * blockDim.x + threadIdx.x;
  if (i < n) atomicAdd(&cnt[batch[i]], 1.0f);
}

__global__ void k_pooled(const float* __restrict__ sums, const float* __restrict__ cnt,
                         _Float16* __restrict__ P, int F, int total) {
  int idx = blockIdx.x * blockDim.x + threadIdx.x;
  if (idx >= total) return;
  int g = idx / F;
  P[idx] = (_Float16)(sums[idx] / fmaxf(cnt[g], 1.0f));
}

// ---------------------------------------------------------------------------
// Host orchestration
// ---------------------------------------------------------------------------
static inline int nblk(long long n) { return (int)((n + 255) / 256); }

extern "C" void kernel_launch(void* const* d_in, const int* in_sizes, int n_in,
                              void* d_out, int out_size, void* d_ws, size_t ws_size,
                              hipStream_t stream) {
  const float* x        = (const float*)d_in[0];
  const float* edge_attr= (const float*)d_in[1];
  const int*   ei       = (const int*)d_in[2];
  const int*   batch    = (const int*)d_in[3];
  const float* emb_w1   = (const float*)d_in[4];  const float* emb_b1 = (const float*)d_in[5];
  const float* emb_w2   = (const float*)d_in[6];  const float* emb_b2 = (const float*)d_in[7];
  const float* ep_w1    = (const float*)d_in[8];  const float* ep_b1  = (const float*)d_in[9];
  const float* ep_w2    = (const float*)d_in[10]; const float* ep_b2  = (const float*)d_in[11];
  const float* c6_w     = (const float*)d_in[12]; const float* c6_b   = (const float*)d_in[13];
  const float* c7_w     = (const float*)d_in[14]; const float* c7_b   = (const float*)d_in[15];
  const float* c8_w     = (const float*)d_in[16]; const float* c8_b   = (const float*)d_in[17];
  const float* fc1_w    = (const float*)d_in[18]; const float* fc1_b  = (const float*)d_in[19];
  const float* head_w   = (const float*)d_in[20]; const float* head_b = (const float*)d_in[21];

  char* ws = (char*)d_ws;
  size_t off = 0;
  auto alloc = [&](size_t bytes) -> void* {
    void* p = ws + off;
    off = (off + bytes + 255) & ~(size_t)255;
    return p;
  };

  _Float16* Xh   = (_Float16*)alloc((size_t)NN * 64 * 2);
  _Float16* W1h  = (_Float16*)alloc((size_t)64 * 512 * 2);
  _Float16* W2h  = (_Float16*)alloc((size_t)512 * 512 * 2);
  _Float16* C6h  = (_Float16*)alloc((size_t)512 * 1024 * 2);
  _Float16* C7h  = (_Float16*)alloc((size_t)1024 * 1024 * 2);
  _Float16* C8h  = (_Float16*)alloc((size_t)1024 * 1024 * 2);
  _Float16* F1h  = (_Float16*)alloc((size_t)1024 * 2048 * 2);
  _Float16* Hdh  = (_Float16*)alloc((size_t)2048 * 384 * 2);
  float*    ew   = (float*)alloc((size_t)NE * 4);
  float*    deg  = (float*)alloc((size_t)NN * 4);
  float*    dis  = (float*)alloc((size_t)NN * 4);
  _Float16* H0h  = (_Float16*)alloc((size_t)NN * 512 * 2);
  _Float16* H1h  = (_Float16*)alloc((size_t)NN * 512 * 2);
  float*    HW   = (float*)alloc((size_t)NN * 1024 * 4);
  float*    Out  = (float*)alloc((size_t)NN * 1024 * 4);
  _Float16* Ha   = (_Float16*)alloc((size_t)NN * 1024 * 2);
  _Float16* Hb   = (_Float16*)alloc((size_t)NN * 1024 * 2);
  float*    sums = (float*)alloc((size_t)NG * 1024 * 4);
  float*    cnt  = (float*)alloc((size_t)NG * 4);
  _Float16* Ph   = (_Float16*)alloc((size_t)NG * 1024 * 2);
  _Float16* Zh   = (_Float16*)alloc((size_t)NG * 2048 * 2);

  // --- weight / input conversion to f16 (with padding) ---
  k_convert_pad<<<nblk((long long)NN * 64), 256, 0, stream>>>(x, Xh, NN, 40, NN, 64);
  k_convert_pad<<<nblk(64 * 512), 256, 0, stream>>>(emb_w1, W1h, 40, 512, 64, 512);
  k_convert_pad<<<nblk(512 * 512), 256, 0, stream>>>(emb_w2, W2h, 512, 512, 512, 512);
  k_convert_pad<<<nblk(512 * 1024), 256, 0, stream>>>(c6_w, C6h, 512, 1024, 512, 1024);
  k_convert_pad<<<nblk(1024 * 1024), 256, 0, stream>>>(c7_w, C7h, 1024, 1024, 1024, 1024);
  k_convert_pad<<<nblk(1024 * 1024), 256, 0, stream>>>(c8_w, C8h, 1024, 1024, 1024, 1024);
  k_convert_pad<<<nblk(1024 * 2048), 256, 0, stream>>>(fc1_w, F1h, 1024, 2048, 1024, 2048);
  k_convert_pad<<<nblk(2048 * 384), 256, 0, stream>>>(head_w, Hdh, 2048, 345, 2048, 384);

  // --- edge weights + normalization ---
  k_edge_mlp<<<nblk(NE), 256, 0, stream>>>(edge_attr, ep_w1, ep_b1, ep_w2, ep_b2, ew, NE);
  k_fill_f32<<<nblk(NN), 256, 0, stream>>>(deg, 1.0f, NN);
  k_deg_edges<<<nblk(NE), 256, 0, stream>>>(ei, ew, deg, NE);
  k_rsqrt<<<nblk(NN), 256, 0, stream>>>(deg, dis, NN);

  const int mb = (NN + 127) / 128;

  // --- embedding MLP ---
  k_wmma_gemm<1, 1, 1><<<dim3(512 / 128, mb), 256, 0, stream>>>(
      Xh, W1h, emb_b1, H0h, nullptr, NN, 64, 512, 512);
  k_wmma_gemm<1, 1, 1><<<dim3(512 / 128, mb), 256, 0, stream>>>(
      H0h, W2h, emb_b2, H1h, nullptr, NN, 512, 512, 512);

  // --- conv 6: 512 -> 1024 ---
  k_wmma_gemm<0, 0, 0><<<dim3(1024 / 128, mb), 256, 0, stream>>>(
      H1h, C6h, nullptr, nullptr, HW, NN, 512, 1024, 1024);
  k_selfloop<<<nblk((long long)NN * 1024), 256, 0, stream>>>(HW, dis, Out, 1024, (long long)NN * 1024);
  k_edge_scatter<<<NE, 256, 0, stream>>>(ei, ew, dis, HW, Out, NE, 1024);
  k_bias_relu_h<<<nblk((long long)NN * 1024), 256, 0, stream>>>(Out, c6_b, Ha, 1024, (long long)NN * 1024);

  // --- conv 7: 1024 -> 1024 ---
  k_wmma_gemm<0, 0, 0><<<dim3(1024 / 128, mb), 256, 0, stream>>>(
      Ha, C7h, nullptr, nullptr, HW, NN, 1024, 1024, 1024);
  k_selfloop<<<nblk((long long)NN * 1024), 256, 0, stream>>>(HW, dis, Out, 1024, (long long)NN * 1024);
  k_edge_scatter<<<NE, 256, 0, stream>>>(ei, ew, dis, HW, Out, NE, 1024);
  k_bias_relu_h<<<nblk((long long)NN * 1024), 256, 0, stream>>>(Out, c7_b, Hb, 1024, (long long)NN * 1024);

  // --- conv 8: 1024 -> 1024 ---
  k_wmma_gemm<0, 0, 0><<<dim3(1024 / 128, mb), 256, 0, stream>>>(
      Hb, C8h, nullptr, nullptr, HW, NN, 1024, 1024, 1024);
  k_selfloop<<<nblk((long long)NN * 1024), 256, 0, stream>>>(HW, dis, Out, 1024, (long long)NN * 1024);
  k_edge_scatter<<<NE, 256, 0, stream>>>(ei, ew, dis, HW, Out, NE, 1024);
  k_bias_relu_h<<<nblk((long long)NN * 1024), 256, 0, stream>>>(Out, c8_b, Ha, 1024, (long long)NN * 1024);

  // --- mean pool over graphs ---
  k_fill_f32<<<nblk(NG * 1024), 256, 0, stream>>>(sums, 0.0f, NG * 1024);
  k_fill_f32<<<1, 256, 0, stream>>>(cnt, 0.0f, NG);
  k_pool_add<<<nblk((long long)NN * 1024), 256, 0, stream>>>(Ha, batch, sums, 1024, (long long)NN * 1024);
  k_count<<<nblk(NN), 256, 0, stream>>>(batch, cnt, NN);
  k_pooled<<<nblk(NG * 1024), 256, 0, stream>>>(sums, cnt, Ph, 1024, NG * 1024);

  // --- fc1 + head ---
  k_wmma_gemm<1, 1, 1><<<dim3(2048 / 128, 1), 256, 0, stream>>>(
      Ph, F1h, fc1_b, Zh, nullptr, NG, 1024, 2048, 2048);
  k_wmma_gemm<0, 0, 1><<<dim3(384 / 128, 1), 256, 0, stream>>>(
      Zh, Hdh, head_b, nullptr, (float*)d_out, NG, 2048, 384, 345);
}
